// GraniteMoeHybridMoEBlock_56650618635048
// MI455X (gfx1250) — matrix operand order repaired and checked by reading, not compiled, not run on hardware
//
#include <hip/hip_runtime.h>
#include <hip/hip_bf16.h>
#include <stdint.h>

// ---------------------------------------------------------------------------
// GraniteMoeHybrid MoE block for MI455X (gfx1250, wave32, WMMA + TDM).
// T=4096 tokens, H=2048, I=1024, E=16, top-2 routing.
// ---------------------------------------------------------------------------

constexpr int T = 4096;
constexpr int H = 2048;
constexpr int I = 1024;
constexpr int E = 16;

typedef _Float16 v16h __attribute__((ext_vector_type(16)));
typedef _Float16 v8h  __attribute__((ext_vector_type(8)));
typedef _Float16 v4h  __attribute__((ext_vector_type(4)));
typedef float    v8f  __attribute__((ext_vector_type(8)));
typedef unsigned int u32x4 __attribute__((ext_vector_type(4)));
typedef int      i32x4 __attribute__((ext_vector_type(4)));
typedef int      i32x8 __attribute__((ext_vector_type(8)));

// ---------------------------------------------------------------------------
// WMMA fragment loaders per CDNA5 ISA wave32 VGPR layouts (§7.12.2).
// Tiles are row-major in LDS with row stride 64 halfs -> every load is a
// 16-byte ds_load_b128.
// ---------------------------------------------------------------------------
__device__ __forceinline__ v16h frag_a(const _Float16* sA, int kk, int lane) {
  int m   = lane & 15;
  int klo = (lane < 16) ? 0 : 8;
  v8h lo = *(const v8h*)(sA + m * 64 + kk + klo);
  v8h hi = *(const v8h*)(sA + m * 64 + kk + 16 + klo);
  v16h r;
#pragma unroll
  for (int i = 0; i < 8; ++i) { r[i] = lo[i]; r[i + 8] = hi[i]; }
  return r;
}

__device__ __forceinline__ v16h frag_b(const _Float16* sB, int nbase, int kk, int lane) {
  int n    = nbase + (lane & 15);
  int koff = (lane < 16) ? 0 : 8;
  v8h lo = *(const v8h*)(sB + n * 64 + kk + koff);
  v8h hi = *(const v8h*)(sB + n * 64 + kk + koff + 16);
  v16h r;
#pragma unroll
  for (int i = 0; i < 8; ++i) { r[i] = lo[i]; r[i + 8] = hi[i]; }
  return r;
}

// ---------------------------------------------------------------------------
// TDM: async 2D f16 tile load (rows x cols) from global into LDS.
// D# built per CDNA5 ISA §8 (group0: count/lds/global addr/type=2;
// group1: data_size=2B, dims, tile dims, dim0 stride). 2D tensor -> groups
// 2/3 zero. Issued by one wave; fenced with s_wait_tensorcnt.
// ---------------------------------------------------------------------------
__device__ __forceinline__ void tdm_load_2d_f16(unsigned lds_off, const void* gptr,
                                                unsigned tensor_d0, unsigned tensor_d1,
                                                unsigned tile_d0, unsigned tile_d1,
                                                unsigned stride0) {
  unsigned long long ga = (unsigned long long)(uintptr_t)gptr;
  u32x4 g0;
  g0[0] = 1u;                                            // count=1, user desc
  g0[1] = lds_off;                                       // LDS byte address
  g0[2] = (unsigned)(ga & 0xFFFFFFFFu);                  // global_addr[31:0]
  g0[3] = (unsigned)((ga >> 32) & 0x1FFFFFFu) | (2u << 30); // [56:32] | type=2
  i32x8 g1;
  g1[0] = (1 << 16);                                     // data_size=1 (2 bytes)
  g1[1] = (int)((tensor_d0 & 0xFFFFu) << 16);            // tensor_dim0[15:0]
  g1[2] = (int)(((tensor_d0 >> 16) & 0xFFFFu) | ((tensor_d1 & 0xFFFFu) << 16));
  g1[3] = (int)(((tensor_d1 >> 16) & 0xFFFFu) | ((tile_d0 & 0xFFFFu) << 16));
  g1[4] = (int)(tile_d1 & 0xFFFFu);                      // tile_dim1, tile_dim2=0
  g1[5] = (int)stride0;                                  // tensor_dim0_stride[31:0]
  g1[6] = 0;
  g1[7] = 0;
  i32x4 gz = {0, 0, 0, 0};
#if defined(__clang_major__) && (__clang_major__ >= 23)
  i32x8 gz8 = {0, 0, 0, 0, 0, 0, 0, 0};
  __builtin_amdgcn_tensor_load_to_lds(g0, g1, gz, gz, gz8, 0);
#else
  __builtin_amdgcn_tensor_load_to_lds(g0, g1, gz, gz, 0);
#endif
}

// ---------------------------------------------------------------------------
// Small utility kernels
// ---------------------------------------------------------------------------
__global__ void k_init_cnt(int* cnt) {
  if (threadIdx.x < E) cnt[threadIdx.x] = 0;
}

__global__ void k_x_to_half(const float* __restrict__ x, _Float16* __restrict__ xh) {
  size_t i = ((size_t)blockIdx.x * blockDim.x + threadIdx.x) * 4;
  float4 v = *(const float4*)(x + i);
  v4h h;
  h[0] = (_Float16)v.x; h[1] = (_Float16)v.y;
  h[2] = (_Float16)v.z; h[3] = (_Float16)v.w;
  *(v4h*)(xh + i) = h;
}

// Router: one wave per token, fp32 math so top-2 selection matches reference.
__global__ void k_router(const float* __restrict__ x, const float* __restrict__ rw,
                         int* __restrict__ topk_idx, float* __restrict__ topk_w,
                         int* __restrict__ cnt) {
  int wave = threadIdx.x >> 5;
  int lane = threadIdx.x & 31;
  int t = blockIdx.x * 8 + wave;

  float acc[E];
#pragma unroll
  for (int e = 0; e < E; ++e) acc[e] = 0.f;

  const float* xt = x + (size_t)t * H;
  for (int k = lane; k < H; k += 32) {
    float xv = xt[k];
#pragma unroll
    for (int e = 0; e < E; ++e) acc[e] += xv * rw[e * H + k];
  }
#pragma unroll
  for (int e = 0; e < E; ++e) {
#pragma unroll
    for (int m = 16; m >= 1; m >>= 1) acc[e] += __shfl_xor(acc[e], m, 32);
  }
  if (lane == 0) {
    float best = -3.4e38f, sec = -3.4e38f;
    int bi = 0, si = 0;
#pragma unroll
    for (int e = 0; e < E; ++e) {
      float v = acc[e];
      if (v > best)     { sec = best; si = bi; best = v; bi = e; }
      else if (v > sec) { sec = v; si = e; }
    }
    float w1  = __expf(sec - best);
    float inv = __builtin_amdgcn_rcpf(1.f + w1);
    topk_idx[2 * t]     = bi;
    topk_idx[2 * t + 1] = si;
    topk_w[2 * t]       = inv;
    topk_w[2 * t + 1]   = w1 * inv;
    atomicAdd(&cnt[bi], 1);
    atomicAdd(&cnt[si], 1);
  }
}

__global__ void k_offsets(const int* __restrict__ cnt, int* __restrict__ off) {
  int s = 0;
  for (int e = 0; e < E; ++e) { off[e] = s; s += cnt[e]; }
}

// Deterministic per-expert compaction (order = ascending token id).
__global__ void k_scatter(const int* __restrict__ topk_idx, const float* __restrict__ topk_w,
                          const int* __restrict__ off,
                          int* __restrict__ tok_list, float* __restrict__ tok_w,
                          int* __restrict__ pair_pos) {
  const int e   = blockIdx.x;
  const int tid = threadIdx.x;
  __shared__ int scan[256];
  int base = off[e];

  for (int c = 0; c < T; c += 256) {
    int t  = c + tid;
    int i0 = topk_idx[2 * t];
    int i1 = topk_idx[2 * t + 1];
    int hit0 = (i0 == e);
    int hit1 = (i1 == e);
    int p = hit0 | hit1;

    scan[tid] = p;
    __syncthreads();
    for (int s = 1; s < 256; s <<= 1) {
      int v = (tid >= s) ? scan[tid - s] : 0;
      __syncthreads();
      scan[tid] += v;
      __syncthreads();
    }
    int excl  = scan[tid] - p;
    int total = scan[255];
    if (p) {
      int   slot = hit0 ? 0 : 1;
      float w    = hit0 ? topk_w[2 * t] : topk_w[2 * t + 1];
      int   pos  = base + excl;
      tok_list[pos] = t;
      tok_w[pos]    = w;
      pair_pos[2 * t + slot] = pos;
    }
    base += total;
    __syncthreads();
  }
}

// ---------------------------------------------------------------------------
// Stage 1: hidden[p, i] = w_pair * silu(x·Wg^T) * (x·Wu^T), f16 out.
// Block = 256 threads (8 waves). Tile: M=32 pairs x N=128 I-cols, K chunks 64.
// Each wave: two A-fragments reuse each B-fragment -> 4 WMMA / 8 ds_load_b128.
// ---------------------------------------------------------------------------
__global__ void __launch_bounds__(256)
k_gate_up(const _Float16* __restrict__ xh,
          const float* __restrict__ w_gate, const float* __restrict__ w_up,
          const int* __restrict__ tok_list, const float* __restrict__ tok_w,
          const int* __restrict__ cnt, const int* __restrict__ off,
          _Float16* __restrict__ hidden) {
  const int e   = blockIdx.z;
  const int n_e = cnt[e];
  const int mt  = blockIdx.y;
  if (mt * 32 >= n_e) return;
  const int n0   = blockIdx.x * 128;       // I offset
  const int tid  = threadIdx.x;
  const int lane = tid & 31;
  const int wave = tid >> 5;

  // 4KB A | 16KB Bg | 16KB Bu | 128B w | 128B tok ; epilogue aliases Bg.
  __shared__ __align__(16) unsigned char smem[4096 + 16384 + 16384 + 256];
  _Float16* sA  = (_Float16*)smem;
  _Float16* sBg = (_Float16*)(smem + 4096);
  _Float16* sBu = (_Float16*)(smem + 4096 + 16384);
  float*    sW   = (float*)(smem + 4096 + 16384 + 16384);
  int*      sTok = (int*)(smem + 4096 + 16384 + 16384 + 128);

  if (tid < 32) {
    int r  = mt * 32 + tid;
    int ok = r < n_e;
    int p  = off[e] + r;
    sTok[tid] = ok ? tok_list[p] : 0;
    sW[tid]   = ok ? tok_w[p]    : 0.f;
  }

  const float* wg = w_gate + (size_t)e * I * H;
  const float* wu = w_up   + (size_t)e * I * H;

  v8f accg[2] = {{}, {}};
  v8f accu[2] = {{}, {}};

  const int arow = tid >> 3;          // 0..31 (A tile rows)
  const int ac8  = (tid & 7) * 8;     // 0..56
  const int brow = tid >> 4;          // 0..15 (B tile base rows)
  const int bc4  = (tid & 15) * 4;    // 0..60

  for (int k0 = 0; k0 < H; k0 += 64) {
    __syncthreads();
    // A tile: 32 gathered f16 token rows (16B loads).
    *(v8h*)(sA + arow * 64 + ac8) =
        *(const v8h*)(xh + (size_t)sTok[arow] * H + k0 + ac8);
    // B tiles: fp32 weights -> f16, 128x64 each.
#pragma unroll
    for (int rr = 0; rr < 8; ++rr) {
      int row = brow + rr * 16;
      float4 g4 = *(const float4*)(wg + (size_t)(n0 + row) * H + k0 + bc4);
      float4 u4 = *(const float4*)(wu + (size_t)(n0 + row) * H + k0 + bc4);
      v4h hg, hu;
      hg[0] = (_Float16)g4.x; hg[1] = (_Float16)g4.y;
      hg[2] = (_Float16)g4.z; hg[3] = (_Float16)g4.w;
      hu[0] = (_Float16)u4.x; hu[1] = (_Float16)u4.y;
      hu[2] = (_Float16)u4.z; hu[3] = (_Float16)u4.w;
      *(v4h*)(sBg + row * 64 + bc4) = hg;
      *(v4h*)(sBu + row * 64 + bc4) = hu;
    }
    __syncthreads();
#pragma unroll
    for (int kk = 0; kk < 64; kk += 32) {
      v16h a0 = frag_a(sA,           kk, lane);
      v16h a1 = frag_a(sA + 16 * 64, kk, lane);
      v16h bg = frag_b(sBg, wave * 16, kk, lane);
      v16h bu = frag_b(sBu, wave * 16, kk, lane);
      accg[0] = __builtin_amdgcn_wmma_f32_16x16x32_f16(false, a0, false, bg,
                                                       (short)0, accg[0], false, false);
      accg[1] = __builtin_amdgcn_wmma_f32_16x16x32_f16(false, a1, false, bg,
                                                       (short)0, accg[1], false, false);
      accu[0] = __builtin_amdgcn_wmma_f32_16x16x32_f16(false, a0, false, bu,
                                                       (short)0, accu[0], false, false);
      accu[1] = __builtin_amdgcn_wmma_f32_16x16x32_f16(false, a1, false, bu,
                                                       (short)0, accu[1], false, false);
    }
  }

  // Epilogue: silu(g)*u scaled by routing weight, staged via LDS so global
  // stores are coalesced b128.
  __syncthreads();
  _Float16* sOut = (_Float16*)(smem + 4096);  // alias Bg: 32x128 f16 = 8KB
  const int colL  = wave * 16 + (lane & 15);
  const int rbase = (lane < 16) ? 0 : 8;
#pragma unroll
  for (int hM = 0; hM < 2; ++hM) {
#pragma unroll
    for (int r = 0; r < 8; ++r) {
      int row = hM * 16 + rbase + r;
      float g = accg[hM][r], u = accu[hM][r];
      float hv = g * __builtin_amdgcn_rcpf(1.f + __expf(-g)) * u * sW[row];
      sOut[row * 128 + colL] = (_Float16)hv;
    }
  }
  __syncthreads();
  {
    int row = tid >> 3;          // 0..31
    int s   = tid & 7;           // 8 threads per row, 16 halfs each
    if (mt * 32 + row < n_e) {
      _Float16* dst = hidden + (size_t)(off[e] + mt * 32 + row) * I + n0;
      *(v8h*)(dst + s * 16)     = *(v8h*)(sOut + row * 128 + s * 16);
      *(v8h*)(dst + s * 16 + 8) = *(v8h*)(sOut + row * 128 + s * 16 + 8);
    }
  }
}

// ---------------------------------------------------------------------------
// Stage 2: Y[p, h] = hidden[p,:] · w_down[e][h,:]  (f32 out).
// Tile: M=32 pairs x N=128 H-cols, K=I in chunks of 64.
// A-tile is fetched by the Tensor Data Mover (contiguous f16 rows).
// ---------------------------------------------------------------------------
__global__ void __launch_bounds__(256)
k_down(const _Float16* __restrict__ hidden, const float* __restrict__ w_down,
       const int* __restrict__ cnt, const int* __restrict__ off,
       float* __restrict__ Y) {
  const int e   = blockIdx.z;
  const int n_e = cnt[e];
  const int mt  = blockIdx.y;
  if (mt * 32 >= n_e) return;
  const int h0   = blockIdx.x * 128;       // H offset
  const int tid  = threadIdx.x;
  const int lane = tid & 31;
  const int wave = tid >> 5;

  // 4KB A | 16KB B ; epilogue (32x128 f32 = 16KB) aliases B.
  __shared__ __align__(16) unsigned char smem[4096 + 16384];
  _Float16* sA = (_Float16*)smem;
  _Float16* sB = (_Float16*)(smem + 4096);

  const float* wd = w_down + (size_t)e * H * I;
  const size_t pbase = (size_t)off[e] + (size_t)mt * 32;
  const unsigned rows_avail = (unsigned)(2 * T + 32 - (int)pbase);
  const unsigned sA_off = (unsigned)(uintptr_t)sA;  // LDS byte offset

  v8f acc[2] = {{}, {}};
  const int brow = tid >> 4;
  const int bc4  = (tid & 15) * 4;

  for (int k0 = 0; k0 < I; k0 += 64) {
    __syncthreads();
    // Async TDM: 32 rows x 64 halfs of `hidden` into sA (row stride I).
    if (wave == 0) {
      tdm_load_2d_f16(sA_off, hidden + pbase * I + k0,
                      (unsigned)(I - k0), rows_avail, 64u, 32u, (unsigned)I);
    }
    // B tile: fp32 w_down -> f16, 128x64.
#pragma unroll
    for (int rr = 0; rr < 8; ++rr) {
      int row = brow + rr * 16;
      float4 d4 = *(const float4*)(wd + (size_t)(h0 + row) * I + k0 + bc4);
      v4h hd;
      hd[0] = (_Float16)d4.x; hd[1] = (_Float16)d4.y;
      hd[2] = (_Float16)d4.z; hd[3] = (_Float16)d4.w;
      *(v4h*)(sB + row * 64 + bc4) = hd;
    }
    if (wave == 0) __builtin_amdgcn_s_wait_tensorcnt(0);
    __syncthreads();
#pragma unroll
    for (int kk = 0; kk < 64; kk += 32) {
      v16h a0 = frag_a(sA,           kk, lane);
      v16h a1 = frag_a(sA + 16 * 64, kk, lane);
      v16h b  = frag_b(sB, wave * 16, kk, lane);
      acc[0] = __builtin_amdgcn_wmma_f32_16x16x32_f16(false, a0, false, b,
                                                      (short)0, acc[0], false, false);
      acc[1] = __builtin_amdgcn_wmma_f32_16x16x32_f16(false, a1, false, b,
                                                      (short)0, acc[1], false, false);
    }
  }

  // Epilogue staged via LDS for coalesced b128 stores.
  __syncthreads();
  float* sOut = (float*)(smem + 4096);  // alias B: 32x128 f32
  const int colL  = wave * 16 + (lane & 15);
  const int rbase = (lane < 16) ? 0 : 8;
#pragma unroll
  for (int hM = 0; hM < 2; ++hM) {
#pragma unroll
    for (int r = 0; r < 8; ++r) {
      int row = hM * 16 + rbase + r;
      sOut[row * 128 + colL] = acc[hM][r];
    }
  }
  __syncthreads();
  {
    int row = tid >> 3;          // 0..31
    int s   = tid & 7;           // 8 threads per row, 4 float4 each
    if (mt * 32 + row < n_e) {
      float* dst = Y + (pbase + row) * H + h0;
#pragma unroll
      for (int j = 0; j < 4; ++j) {
        int u = s + 8 * j;       // contiguous 128B per 8-thread group
        *(float4*)(dst + u * 4) = *(float4*)(sOut + row * 128 + u * 4);
      }
    }
  }
}

// out[t] = Y[pair0(t)] + Y[pair1(t)]  (routing weights already folded in).
__global__ void k_combine(const float* __restrict__ Y, const int* __restrict__ pair_pos,
                          float* __restrict__ out) {
  size_t i = (size_t)blockIdx.x * blockDim.x + threadIdx.x;  // float4 units
  size_t t = i / (H / 4);
  size_t c = i % (H / 4);
  const float4* Y4 = (const float4*)Y;
  float4 a = Y4[(size_t)pair_pos[2 * t] * (H / 4) + c];
  float4 b = Y4[(size_t)pair_pos[2 * t + 1] * (H / 4) + c];
  float4 r;
  r.x = a.x + b.x; r.y = a.y + b.y; r.z = a.z + b.z; r.w = a.w + b.w;
  ((float4*)out)[i] = r;
}

// ---------------------------------------------------------------------------
extern "C" void kernel_launch(void* const* d_in, const int* in_sizes, int n_in,
                              void* d_out, int out_size, void* d_ws, size_t ws_size,
                              hipStream_t stream) {
  const float* x      = (const float*)d_in[0];  // [T, H]
  const float* rw     = (const float*)d_in[1];  // [E, H]
  const float* w_gate = (const float*)d_in[2];  // [E, I, H]
  const float* w_up   = (const float*)d_in[3];  // [E, I, H]
  const float* w_down = (const float*)d_in[4];  // [E, H, I]
  float* out = (float*)d_out;

  char*  p    = (char*)d_ws;
  size_t used = 0;
  auto alloc = [&](size_t bytes) -> void* {
    void* r = p;
    size_t a = (bytes + 255) & ~(size_t)255;
    p += a; used += a;
    return r;
  };
  _Float16* xh       = (_Float16*)alloc((size_t)T * H * 2);
  _Float16* hidden   = (_Float16*)alloc((size_t)(2 * T + 32) * I * 2); // +32 pad rows
  float*    Y        = (float*)   alloc((size_t)2 * T * H * 4);
  int*      topk_idx = (int*)     alloc((size_t)T * 2 * 4);
  float*    topk_w   = (float*)   alloc((size_t)T * 2 * 4);
  int*      pair_pos = (int*)     alloc((size_t)T * 2 * 4);
  int*      tok_list = (int*)     alloc((size_t)2 * T * 4);
  float*    tok_w    = (float*)   alloc((size_t)2 * T * 4);
  int*      cnt      = (int*)     alloc((size_t)E * 4);
  int*      off      = (int*)     alloc((size_t)E * 4);
  if (used > ws_size) return;  // insufficient scratch: bail deterministically

  k_init_cnt<<<1, 32, 0, stream>>>(cnt);
  k_x_to_half<<<(T * H) / (256 * 4), 256, 0, stream>>>(x, xh);
  k_router<<<T / 8, 256, 0, stream>>>(x, rw, topk_idx, topk_w, cnt);
  k_offsets<<<1, 1, 0, stream>>>(cnt, off);
  k_scatter<<<E, 256, 0, stream>>>(topk_idx, topk_w, off, tok_list, tok_w, pair_pos);

  dim3 g1(I / 128, 128, E);  // (N tiles, max M tiles of 32, experts)
  k_gate_up<<<g1, 256, 0, stream>>>(xh, w_gate, w_up, tok_list, tok_w, cnt, off, hidden);

  dim3 g2(H / 128, 128, E);
  k_down<<<g2, 256, 0, stream>>>(hidden, w_down, cnt, off, Y);

  k_combine<<<(T * (H / 4)) / 256, 256, 0, stream>>>(Y, pair_pos, out);
}
